// Transformer_10565619548190
// MI455X (gfx1250) — matrix operand order repaired and checked by reading, not compiled, not run on hardware
//
#include <hip/hip_runtime.h>
#include <math.h>

// ---------------------------------------------------------------------------
// Transformer forward (2 layers) for MI455X (gfx1250, wave32, WMMA).
// All matmuls on v_wmma_f32_16x16x32_bf16 (bf16 in, f32 accumulate).
// Block tile 128x128, 8 waves (4Mx2N), wave tile 32x64, K-step 32.
// Global->LDS staging uses GLOBAL_LOAD_ASYNC_TO_LDS_B128 (ASYNCcnt) with
// double-buffered LDS; all B operands stored n-major (transposed).
// ---------------------------------------------------------------------------

typedef unsigned short u16;
typedef __attribute__((ext_vector_type(16))) __bf16      v16bf;
typedef __attribute__((ext_vector_type(8)))  float       v8f;
typedef __attribute__((ext_vector_type(4)))  unsigned int u32x4;

#define BB   8
#define SS   1024
#define DD   1024
#define HH   8
#define DKK  128
#define DFFF 2048
#define NLAY 2

#define LDSTRIDE 40          // 32 K-elems + 8 pad (u16 units)
#define BUFSZ (128 * LDSTRIDE)

union FragU { u32x4 u[2]; v16bf v; };

__device__ __forceinline__ u16 f2bf(float f) {
  union { float f; unsigned int u; } c; c.f = f;
  unsigned int r = c.u + 0x7FFFu + ((c.u >> 16) & 1u); // round-to-nearest-even
  return (u16)(r >> 16);
}

// Async global->LDS 16B copy (CDNA5 TDM-class async op, tracked by ASYNCcnt).
// ldsOff: byte address in LDS (AS3 offset = low 32 bits of flat LDS pointer).
__device__ __forceinline__ void async_g2l_b128(unsigned ldsOff, const u16* g) {
  asm volatile("global_load_async_to_lds_b128 %0, %1, off"
               :: "v"(ldsOff), "v"((unsigned long long)(size_t)g)
               : "memory");
}
__device__ __forceinline__ void wait_async0() {
  asm volatile("s_wait_asynccnt 0" ::: "memory");
}

// ---------------------------------------------------------------------------
// C = A(MxK) * B(KxN) [+bias] [relu].
// A: row-major bf16, lda.  Bm: n-major bf16 (element [k,n] = Bm[n*ldb+k]).
// Output: OF32 -> Cf row-major f32; OB16 -> Cb row-major bf16;
//         OB16T -> Cb col-major bf16 (Cb[col*ldc + row], 16B stores).
// grid = (N/128, M/128, batch); sA/sB/sC batch strides in elements.
// ---------------------------------------------------------------------------
template<bool HAS_BIAS, bool DO_RELU, bool OF32, bool OB16, bool OB16T>
__global__ __launch_bounds__(256) void gemm_bf16_wmma(
    const u16* __restrict__ A, int lda, long long sA,
    const u16* __restrict__ Bm, int ldb, long long sB,
    float* __restrict__ Cf, u16* __restrict__ Cb, int ldc, long long sC,
    int K, const float* __restrict__ bias)
{
  __shared__ __align__(16) u16 As[2 * BUFSZ];   // [buf][row 0..127][k 0..31 +pad]
  __shared__ __align__(16) u16 Bs[2 * BUFSZ];   // [buf][col 0..127][k 0..31 +pad]

  const int tid  = threadIdx.x;
  const int lane = tid & 31;
  const int wave = tid >> 5;
  const int wM   = wave >> 1;          // 0..3
  const int wN   = wave & 1;           // 0..1
  const long long bz = blockIdx.z;

  A  += bz * sA;
  Bm += bz * sB;
  const int rowBlk = blockIdx.y * 128;
  const int colBlk = blockIdx.x * 128;

  // staging map: 128 rows x 32 k, 16 u16 per thread (two 16B chunks)
  const int sr = tid >> 1;             // 0..127  (row of A / col of B)
  const int sk = (tid & 1) << 4;       // 0 or 16 (k offset)
  const u16* Aga = A  + (size_t)(rowBlk + sr) * lda + sk;
  const u16* Bga = Bm + (size_t)(colBlk + sr) * ldb + sk;

  // LDS byte addresses for this thread's staging slots
  const unsigned ldsA = (unsigned)(size_t)&As[sr * LDSTRIDE + sk];
  const unsigned ldsB = (unsigned)(size_t)&Bs[sr * LDSTRIDE + sk];

  v8f acc[2][4] = {};
  const int nk = K >> 5;

  // ---- prologue: async-stage tile 0 into buffer 0 ----
  async_g2l_b128(ldsA,      Aga);
  async_g2l_b128(ldsA + 16, Aga + 8);
  async_g2l_b128(ldsB,      Bga);
  async_g2l_b128(ldsB + 16, Bga + 8);

  const int khalf = lane >> 4;      // A frag: lanes 16-31 hold K+8 halves
  const int frow  = lane & 15;

  int p = 0;
  for (int kt = 0; kt < nk; ++kt) {
    wait_async0();            // this wave's LDS writes for buf p are done
    __syncthreads();          // everyone's are

    // ---- async-stage next K-tile into the other buffer (overlaps WMMA) ----
    if (kt + 1 < nk) {
      const u16* an = Aga + ((kt + 1) << 5);
      const u16* bn = Bga + ((kt + 1) << 5);
      const unsigned boff = (unsigned)((p ^ 1) * (BUFSZ * 2));
      async_g2l_b128(ldsA + boff,      an);
      async_g2l_b128(ldsA + boff + 16, an + 8);
      async_g2l_b128(ldsB + boff,      bn);
      async_g2l_b128(ldsB + boff + 16, bn + 8);
    }

    const u16* Ab = &As[p * BUFSZ];
    const u16* Bb = &Bs[p * BUFSZ];

    // ---- fragment loads (ISA 7.12.2 layouts) ----
    FragU fa[2], fb[4];
#pragma unroll
    for (int m0 = 0; m0 < 2; ++m0) {
      const int base = (wM * 32 + m0 * 16 + frow) * LDSTRIDE + khalf * 8;
      fa[m0].u[0] = *reinterpret_cast<const u32x4*>(&Ab[base]);       // K 0..7 (+8h)
      fa[m0].u[1] = *reinterpret_cast<const u32x4*>(&Ab[base + 16]);  // K 16..23 (+8h)
    }
#pragma unroll
    for (int n0 = 0; n0 < 4; ++n0) {
      // B frag: lane holds col frow, K 0..7 and 16..23 (+8*(lane/16))
      const int base = (wN * 64 + n0 * 16 + frow) * LDSTRIDE + khalf * 8;
      fb[n0].u[0] = *reinterpret_cast<const u32x4*>(&Bb[base]);
      fb[n0].u[1] = *reinterpret_cast<const u32x4*>(&Bb[base + 16]);
    }

    // ---- 8 WMMAs ----
#pragma unroll
    for (int m0 = 0; m0 < 2; ++m0)
#pragma unroll
      for (int n0 = 0; n0 < 4; ++n0)
        acc[m0][n0] = __builtin_amdgcn_wmma_f32_16x16x32_bf16(
            false, fa[m0].v, false, fb[n0].v, (short)0, acc[m0][n0],
            false, false);

    p ^= 1;
  }

  // ---- epilogue ----
  float* CfB = OF32 ? (Cf + bz * sC) : nullptr;
  u16*   CbB = (OB16 || OB16T) ? (Cb + bz * sC) : nullptr;
#pragma unroll
  for (int m0 = 0; m0 < 2; ++m0) {
    const int row0 = rowBlk + wM * 32 + m0 * 16 + (lane >> 4) * 8;
#pragma unroll
    for (int n0 = 0; n0 < 4; ++n0) {
      const int col = colBlk + wN * 64 + n0 * 16 + (lane & 15);
      const float bv = HAS_BIAS ? bias[col] : 0.0f;
      if (OB16T) {
        union { u32x4 u; u16 s[8]; } pk;
#pragma unroll
        for (int r = 0; r < 8; ++r) {
          float v = acc[m0][n0][r] + bv;
          if (DO_RELU) v = fmaxf(v, 0.0f);
          pk.s[r] = f2bf(v);
        }
        *reinterpret_cast<u32x4*>(&CbB[(size_t)col * ldc + row0]) = pk.u;
      } else {
#pragma unroll
        for (int r = 0; r < 8; ++r) {
          float v = acc[m0][n0][r] + bv;
          if (DO_RELU) v = fmaxf(v, 0.0f);
          const size_t off = (size_t)(row0 + r) * ldc + col;
          if (OF32) CfB[off] = v;
          if (OB16) CbB[off] = f2bf(v);
        }
      }
    }
  }
}

// ---------------------------------------------------------------------------
// Masked softmax over rows of length S=1024; one block (8 waves) per row.
// ---------------------------------------------------------------------------
__global__ __launch_bounds__(256) void softmax_mask_k(
    const float* __restrict__ scores, u16* __restrict__ probs,
    const float* __restrict__ mask)
{
  const int row  = blockIdx.x;
  const size_t base = (size_t)row * SS;
  const int tid = threadIdx.x, lane = tid & 31, wave = tid >> 5;
  __shared__ float red[8];

  float v[4];
  float mx = -3.4e38f;
#pragma unroll
  for (int j = 0; j < 4; ++j) {
    const int t = j * 256 + tid;
    const float m = mask[t];
    float s = scores[base + t];
    s = m * s + (1.0f - m) * (-1e-30f);
    v[j] = s;
    mx = fmaxf(mx, s);
  }
#pragma unroll
  for (int off = 16; off; off >>= 1) mx = fmaxf(mx, __shfl_xor(mx, off, 32));
  if (lane == 0) red[wave] = mx;
  __syncthreads();
  mx = red[0];
#pragma unroll
  for (int w = 1; w < 8; ++w) mx = fmaxf(mx, red[w]);
  __syncthreads();

  float sum = 0.0f;
#pragma unroll
  for (int j = 0; j < 4; ++j) { v[j] = __expf(v[j] - mx); sum += v[j]; }
#pragma unroll
  for (int off = 16; off; off >>= 1) sum += __shfl_xor(sum, off, 32);
  if (lane == 0) red[wave] = sum;
  __syncthreads();
  sum = 0.0f;
#pragma unroll
  for (int w = 0; w < 8; ++w) sum += red[w];
  const float inv = 1.0f / sum;
#pragma unroll
  for (int j = 0; j < 4; ++j) probs[base + j * 256 + tid] = f2bf(v[j] * inv);
}

// ---------------------------------------------------------------------------
// Xnew = LayerNorm(Xold + delta) * gamma[gi] + beta[gi]; also bf16 copy.
// ---------------------------------------------------------------------------
__global__ __launch_bounds__(256) void add_ln_k(
    const float* __restrict__ Xold, const float* __restrict__ delta,
    float* __restrict__ Xnew, u16* __restrict__ Xb,
    const float* __restrict__ gamma, const float* __restrict__ beta, int gi)
{
  const int row = blockIdx.x;
  const size_t base = (size_t)row * DD;
  const int tid = threadIdx.x, lane = tid & 31, wave = tid >> 5;
  __shared__ float rs[8], rq[8];

  float x[4];
  float sum = 0.0f, sq = 0.0f;
#pragma unroll
  for (int j = 0; j < 4; ++j) {
    const int c = j * 256 + tid;
    const float t = Xold[base + c] + delta[base + c];
    x[j] = t; sum += t; sq += t * t;
  }
#pragma unroll
  for (int off = 16; off; off >>= 1) {
    sum += __shfl_xor(sum, off, 32);
    sq  += __shfl_xor(sq,  off, 32);
  }
  if (lane == 0) { rs[wave] = sum; rq[wave] = sq; }
  __syncthreads();
  sum = 0.0f; sq = 0.0f;
#pragma unroll
  for (int w = 0; w < 8; ++w) { sum += rs[w]; sq += rq[w]; }

  const float mean = sum * (1.0f / (float)DD);
  const float var  = sq * (1.0f / (float)DD) - mean * mean;
  const float rstd = rsqrtf(var + 1e-14f);       // EPS = (1e-7)^2
  const float g = gamma[gi], b = beta[gi];
#pragma unroll
  for (int j = 0; j < 4; ++j) {
    const int c = j * 256 + tid;
    const float y = (x[j] - mean) * rstd * g + b;
    Xnew[base + c] = y;
    Xb[base + c]   = f2bf(y);
  }
}

// fp32 -> bf16 bulk convert (same layout)
__global__ __launch_bounds__(256) void cvt_bf16_k(
    const float* __restrict__ src, u16* __restrict__ dst, int n)
{
  const int i = blockIdx.x * 256 + threadIdx.x;
  if (i < n) dst[i] = f2bf(src[i]);
}

// fp32 (K x N) row-major -> bf16 transposed (N x K): dst[n*K + k] = src[k*N + n]
__global__ __launch_bounds__(256) void cvt_t_bf16_k(
    const float* __restrict__ src, u16* __restrict__ dst, int Kd, int Nd)
{
  const int i = blockIdx.x * 256 + threadIdx.x;   // over Kd*Nd
  const int n = i / Kd;
  const int k = i - n * Kd;
  dst[i] = f2bf(src[(size_t)k * Nd + n]);
}

// Wq/Wk/Wv (H,D,DK) fp32 -> bf16 transposed (H*DK, D): dst[(h*DK+kk)*D + d]
__global__ __launch_bounds__(256) void pack_hw_t_k(
    const float* __restrict__ src, u16* __restrict__ dst)
{
  const int i = blockIdx.x * 256 + threadIdx.x;   // over H*DK*D = 1M
  const int n  = i >> 10;          // h*DK + kk
  const int d  = i & 1023;
  const int h  = n >> 7;
  const int kk = n & 127;
  dst[i] = f2bf(src[((size_t)h << 17) + ((size_t)d << 7) + kk]); // h*D*DK + d*DK + kk
}

// ---------------------------------------------------------------------------
extern "C" void kernel_launch(void* const* d_in, const int* in_sizes, int n_in,
                              void* d_out, int out_size, void* d_ws, size_t ws_size,
                              hipStream_t stream)
{
  (void)in_sizes; (void)n_in; (void)out_size; (void)ws_size;

  const float* X     = (const float*)d_in[0];
  const float* mask  = (const float*)d_in[1];
  const float* Wq    = (const float*)d_in[2];
  const float* Wk    = (const float*)d_in[3];
  const float* Wv    = (const float*)d_in[4];
  const float* Wo    = (const float*)d_in[5];
  const float* W1    = (const float*)d_in[6];
  const float* b1    = (const float*)d_in[7];
  const float* W2    = (const float*)d_in[8];
  const float* b2    = (const float*)d_in[9];
  const float* gamma = (const float*)d_in[10];
  const float* beta  = (const float*)d_in[11];
  float* Xout = (float*)d_out;

  const size_t BS = (size_t)BB * SS;          // 8192 tokens
  char* ws = (char*)d_ws;
  size_t off = 0;
  auto alloc = [&](size_t bytes) -> void* {
    void* p = ws + off;
    off += (bytes + 255) & ~(size_t)255;
    return p;
  };

  u16*   Xb     = (u16*)  alloc(BS * DD * 2);                 // 16 MB
  u16*   wqT    = (u16*)  alloc((size_t)DD * DD * 2);         //  2 MB (n-major)
  u16*   wkT    = (u16*)  alloc((size_t)DD * DD * 2);
  u16*   wvT    = (u16*)  alloc((size_t)DD * DD * 2);
  u16*   woT    = (u16*)  alloc((size_t)DD * DD * 2);
  u16*   w1T    = (u16*)  alloc((size_t)DD * DFFF * 2);       //  4 MB (n-major)
  u16*   w2T    = (u16*)  alloc((size_t)DFFF * DD * 2);
  u16*   qb     = (u16*)  alloc(BS * DD * 2);                 // 16 MB (b,s,h,dk)
  u16*   kb     = (u16*)  alloc(BS * DD * 2);                 // 16 MB (b,s,h,dk)
  u16*   vT     = (u16*)  alloc(BS * DD * 2);                 // 16 MB (h,dk, b*s)
  u16*   concb  = (u16*)  alloc(BS * DD * 2);
  float* proj   = (float*)alloc(BS * DD * 4);                 // 32 MB
  u16*   h1b    = (u16*)  alloc(BS * DFFF * 2);               // 32 MB
  float* ffn    = (float*)alloc(BS * DD * 4);                 // 32 MB
  float* scores = (float*)alloc((size_t)HH * SS * SS * 4);    // 32 MB (per b)
  u16*   probs  = (u16*)  alloc((size_t)HH * SS * SS * 2);    // 16 MB (per b)

  const dim3 blk(256);

  // X -> bf16
  cvt_bf16_k<<<dim3((unsigned)(BS * DD / 256)), blk, 0, stream>>>(X, Xb, (int)(BS * DD));

  for (int i = 0; i < NLAY; ++i) {
    const size_t wOff  = (size_t)i * DD * DD;
    const size_t w1Off = (size_t)i * DD * DFFF;
    const size_t w2Off = (size_t)i * DFFF * DD;

    // weight repack (all transposed / n-major, bf16)
    pack_hw_t_k<<<4096, blk, 0, stream>>>(Wq + wOff, wqT);
    pack_hw_t_k<<<4096, blk, 0, stream>>>(Wk + wOff, wkT);
    pack_hw_t_k<<<4096, blk, 0, stream>>>(Wv + wOff, wvT);
    cvt_t_bf16_k<<<4096, blk, 0, stream>>>(Wo + wOff,  woT, DD,   DD);
    cvt_t_bf16_k<<<8192, blk, 0, stream>>>(W1 + w1Off, w1T, DD,   DFFF);
    cvt_t_bf16_k<<<8192, blk, 0, stream>>>(W2 + w2Off, w2T, DFFF, DD);

    // Q, K: (8192x1024)x(1024x1024) -> bf16 row-major (b,s,h,dk)
    gemm_bf16_wmma<false,false,false,true,false>
        <<<dim3(8,64,1), blk, 0, stream>>>(Xb,1024,0, wqT,1024,0,
            nullptr, qb, 1024, 0, 1024, nullptr);
    gemm_bf16_wmma<false,false,false,true,false>
        <<<dim3(8,64,1), blk, 0, stream>>>(Xb,1024,0, wkT,1024,0,
            nullptr, kb, 1024, 0, 1024, nullptr);
    // V: same GEMM but store transposed -> vT[n][b*s], n = h*128+dk
    gemm_bf16_wmma<false,false,false,false,true>
        <<<dim3(8,64,1), blk, 0, stream>>>(Xb,1024,0, wvT,1024,0,
            nullptr, vT, 8192, 0, 1024, nullptr);

    // attention, one batch b at a time (grid.z = heads)
    for (int b = 0; b < BB; ++b) {
      const size_t tokOff = (size_t)b * SS * 1024;
      // scores[h,s,t] = q.k : A=q rows, B=k rows (k already n-major: n=t, k=dk)
      gemm_bf16_wmma<false,false,true,false,false>
          <<<dim3(8,8,8), blk, 0, stream>>>(
              qb + tokOff, 1024, 128,
              kb + tokOff, 1024, 128,
              scores, nullptr, 1024, (long long)SS * SS, 128, nullptr);
      // masked softmax -> bf16 probs
      softmax_mask_k<<<dim3(HH * SS), blk, 0, stream>>>(scores, probs, mask + (size_t)b * SS);
      // o[h,s,dk] = probs @ v : B = vT (n-major: n=dk rows of 8192)
      gemm_bf16_wmma<false,false,false,true,false>
          <<<dim3(1,8,8), blk, 0, stream>>>(
              probs, 1024, (long long)SS * SS,
              vT + (size_t)b * SS, 8192, (long long)128 * 8192,
              nullptr, concb + tokOff, 1024, 128, 1024, nullptr);
    }

    // proj = conc @ Wo -> fp32
    gemm_bf16_wmma<false,false,true,false,false>
        <<<dim3(8,64,1), blk, 0, stream>>>(concb,1024,0, woT,1024,0,
            proj, nullptr, 1024, 0, 1024, nullptr);

    // X = LN(X + proj)
    add_ln_k<<<dim3((unsigned)BS), blk, 0, stream>>>(
        (i == 0 ? X : Xout), proj, Xout, Xb, gamma, beta, 2 * i);

    // h1 = relu(X @ W1 + b1) -> bf16 (8192x2048)
    gemm_bf16_wmma<true,true,false,true,false>
        <<<dim3(16,64,1), blk, 0, stream>>>(Xb,1024,0, w1T,1024,0,
            nullptr, h1b, 2048, 0, 1024, b1 + (size_t)i * DFFF);

    // ffn = h1 @ W2 + b2 -> fp32
    gemm_bf16_wmma<true,false,true,false,false>
        <<<dim3(8,64,1), blk, 0, stream>>>(h1b,2048,0, w2T,2048,0,
            ffn, nullptr, 1024, 0, 2048, b2 + (size_t)i * DD);

    // X = LN(X + ffn)
    add_ln_k<<<dim3((unsigned)BS), blk, 0, stream>>>(
        Xout, ffn, Xout, Xb, gamma, beta, 2 * i + 1);
  }
}
